// MultiHeadAttention_56513179681124
// MI455X (gfx1250) — matrix operand order repaired and checked by reading, not compiled, not run on hardware
//
#include <hip/hip_runtime.h>

// ---------------- constants ----------------
constexpr int Bz = 2;           // batch
constexpr int NN = 2048;        // sequence length
constexpr int DD = 1024;        // model dim
constexpr int HH = 16;          // heads
constexpr int HD = 64;          // head dim
constexpr int BN = Bz * NN;     // 4096 rows
constexpr float SCALE = 0.125f; // hd^-0.5

typedef __attribute__((ext_vector_type(16))) _Float16 v16h;
typedef __attribute__((ext_vector_type(8)))  _Float16 v8h;
typedef __attribute__((ext_vector_type(4)))  _Float16 v4h;
typedef __attribute__((ext_vector_type(8)))  float    v8f;

// ---------------- WMMA wrapper ----------------
__device__ __forceinline__ v8f wmma_f16(v16h a, v16h b, v8f c) {
    // D = A(16x32 f16) * B(32x16 f16) + C(16x16 f32)
    return __builtin_amdgcn_wmma_f32_16x16x32_f16(
        /*neg_a=*/false, a, /*neg_b=*/false, b,
        /*c_mod=*/(short)0, c, /*reuse_a=*/false, /*reuse_b=*/false);
}

// ---------------- fragment loaders (contiguous k order, same for A and B) ----------------
// A 16x32 f16: lane&15 = M row; elem e -> K = e + (l>=16?16:0).
//              base = &src[m0*ld + k0].
__device__ __forceinline__ v16h load_afrag(const _Float16* base, int ld, int lane) {
    const _Float16* p = base + (lane & 15) * ld + ((lane >= 16) ? 16 : 0);
    union { v16h v; v8h h[2]; } u;
    u.h[0] = *(const v8h*)p;
    u.h[1] = *(const v8h*)(p + 8);
    return u.v;
}

// B 32x16 f16 from "Bt" storage (row n of src = column n of B, contiguous in K):
// lane&15 = N col; elem e -> K = e + (l>=16?16:0). base = &src[n0*ld + k0].
__device__ __forceinline__ v16h load_bfrag_rm(const _Float16* base, int ld, int lane) {
    const _Float16* p = base + (lane & 15) * ld + ((lane >= 16) ? 16 : 0);
    union { v16h v; v8h h[2]; } u;
    u.h[0] = *(const v8h*)p;
    u.h[1] = *(const v8h*)(p + 8);
    return u.v;
}

// ---------------- kernel 1: x (f32) -> xh (f16), 4 elems/thread ----------------
__global__ __launch_bounds__(256) void cvt_x(const float* __restrict__ x,
                                             _Float16* __restrict__ xh) {
    int i = (blockIdx.x * 256 + threadIdx.x) * 4;
    float4 v = *(const float4*)(x + i);
    v4h o;
    o.x = (_Float16)v.x; o.y = (_Float16)v.y;
    o.z = (_Float16)v.z; o.w = (_Float16)v.w;
    *(v4h*)(xh + i) = o;
}

// ---------------- kernel 2: pack weights into B-fragment layout ----------------
// wp index: ((((mat*32 + kt)*64 + nt)*32 + lane)*16 + e)
//   value = W[mat][n = nt*16 + (lane&15)][k = kt*32 + e + (lane>=16?16:0)]
__global__ __launch_bounds__(256) void pack_w(const float* __restrict__ Wq,
                                              const float* __restrict__ Wk,
                                              const float* __restrict__ Wv,
                                              const float* __restrict__ Wo,
                                              _Float16* __restrict__ wp) {
    int i = blockIdx.x * 256 + threadIdx.x;          // 4*DD*DD threads (2^22)
    int e   =  i        & 15;
    int l   = (i >> 4)  & 31;
    int nt  = (i >> 9)  & 63;
    int kt  = (i >> 15) & 31;
    int mat =  i >> 20;
    const float* Ws[4] = {Wq, Wk, Wv, Wo};
    const float* W = Ws[mat];
    int nrow = nt * 16 + (l & 15);
    int kcol = kt * 32 + e + ((l >= 16) ? 16 : 0);
    wp[i] = (_Float16)W[(size_t)nrow * DD + kcol];
}

// ---------------- kernel 3: fused QKV projection + RoPE ----------------
// grid: x = BN/256 row supertiles, y = 3*HH (mat, head). 8 waves/block,
// wave w owns rows [bx*256 + 32w, +32) x cols [head*64, +64).
// 2x-unrolled ping-pong pipeline: no inter-buffer register copies, and loads
// always write the slot the in-flight WMMAs are NOT reading (no WAR nops).
__global__ __launch_bounds__(256) void qkv_rope(const _Float16* __restrict__ xh,
                                                const _Float16* __restrict__ wp,
                                                _Float16* __restrict__ Qh,
                                                _Float16* __restrict__ Kh,
                                                _Float16* __restrict__ Vt) {
    const int lane = threadIdx.x & 31;
    const int wave = threadIdx.x >> 5;
    const int row0 = blockIdx.x * 256 + wave * 32;   // 0..BN-32
    const int mat  = blockIdx.y >> 4;                // 0=Q 1=K 2=V
    const int head = blockIdx.y & 15;
    const int nt0  = head * 4;                       // first 16-col tile

    const _Float16* ab = xh + (size_t)row0 * DD;
    // B fragment stream: element (kt,t) at base + (kt*64 + t)*512
    const _Float16* wb = wp + ((size_t)mat * 2048 + nt0) * 512 + lane * 16;

    v8f zero = {0.f,0.f,0.f,0.f,0.f,0.f,0.f,0.f};
    v8f c[2][4] = {{zero,zero,zero,zero},{zero,zero,zero,zero}};

    v16h a0[2], a1[2], b[2][4];
    a0[0] = load_afrag(ab, DD, lane);
    a1[0] = load_afrag(ab + 16 * DD, DD, lane);
#pragma unroll
    for (int t = 0; t < 4; ++t) b[0][t] = *(const v16h*)(wb + (size_t)t * 512);

    for (int kt = 0; kt < DD / 32; kt += 2) {
        // stage kt+1 into slot 1 (kt+1 <= 31 always)
        a0[1] = load_afrag(ab + (kt + 1) * 32, DD, lane);
        a1[1] = load_afrag(ab + 16 * DD + (kt + 1) * 32, DD, lane);
#pragma unroll
        for (int t = 0; t < 4; ++t)
            b[1][t] = *(const v16h*)(wb + ((size_t)(kt + 1) * 64 + t) * 512);
        __builtin_prefetch(wb + (((size_t)(kt + 4) & 31) * 64) * 512, 0, 3);
        // compute kt from slot 0
#pragma unroll
        for (int t = 0; t < 4; ++t) {
            c[0][t] = wmma_f16(a0[0], b[0][t], c[0][t]);
            c[1][t] = wmma_f16(a1[0], b[0][t], c[1][t]);
        }
        // stage kt+2 into slot 0 (wrap keeps last iteration defined)
        const int k2 = (kt + 2) & 31;
        a0[0] = load_afrag(ab + k2 * 32, DD, lane);
        a1[0] = load_afrag(ab + 16 * DD + k2 * 32, DD, lane);
#pragma unroll
        for (int t = 0; t < 4; ++t)
            b[0][t] = *(const v16h*)(wb + ((size_t)k2 * 64 + t) * 512);
        // compute kt+1 from slot 1
#pragma unroll
        for (int t = 0; t < 4; ++t) {
            c[0][t] = wmma_f16(a0[1], b[1][t], c[0][t]);
            c[1][t] = wmma_f16(a1[1], b[1][t], c[1][t]);
        }
    }

    // RoPE for Q/K (C layout: lane col = 16t+(lane&15), reg r -> M row)
    if (mat < 2) {
        const float L2_10000_OVER_32 = 13.287712379549449f / 32.f;
        int fr = lane & 15;
        float if0 = exp2f(-(float)fr        * L2_10000_OVER_32); // invfreq[fr]
        float if1 = exp2f(-(float)(fr + 16) * L2_10000_OVER_32); // invfreq[fr+16]
#pragma unroll
        for (int mt = 0; mt < 2; ++mt) {
#pragma unroll
            for (int r = 0; r < 8; ++r) {
                int M = mt * 16 + ((lane < 16) ? r : 8 + r);
                float p = (float)((row0 + M) & (NN - 1));  // position in sequence
                float c0 = __cosf(p * if0), s0 = __sinf(p * if0);
                float c1 = __cosf(p * if1), s1 = __sinf(p * if1);
                float q0 = c[mt][0][r], q1 = c[mt][1][r];
                float q2 = c[mt][2][r], q3 = c[mt][3][r];
                c[mt][0][r] = q0 * c0 - q2 * s0;   // f in [0,16)
                c[mt][1][r] = q1 * c1 - q3 * s1;   // f in [16,32)
                c[mt][2][r] = q2 * c0 + q0 * s0;   // f in [32,48)
                c[mt][3][r] = q3 * c1 + q1 * s1;   // f in [48,64)
            }
        }
    }

    // store: Q/K -> [b][h][n][hd] (row-major per head), V -> [b][h][hd][n]
#pragma unroll
    for (int mt = 0; mt < 2; ++mt) {
#pragma unroll
        for (int r = 0; r < 8; ++r) {
            int M   = mt * 16 + ((lane < 16) ? r : 8 + r);
            int row = row0 + M;
            int bb  = row >> 11;
            int n   = row & (NN - 1);
#pragma unroll
            for (int t = 0; t < 4; ++t) {
                int f = t * 16 + (lane & 15);
                _Float16 val = (_Float16)c[mt][t][r];
                if (mat == 0)
                    Qh[(((size_t)bb * HH + head) * NN + n) * HD + f] = val;
                else if (mat == 1)
                    Kh[(((size_t)bb * HH + head) * NN + n) * HD + f] = val;
                else
                    Vt[(((size_t)bb * HH + head) * HD + f) * NN + n] = val;
            }
        }
    }
}

// ---------------- kernel 4: flash attention (online softmax) ----------------
// grid: x = NN/128 query supertiles, y = B*HH. Wave owns 16 query rows.
// 2x-unrolled ping-pong on the 32-key blocks: next block's K fragments load
// into the idle slot while this block runs S-WMMA + softmax + PV-WMMA.
__global__ __launch_bounds__(256) void attn(const _Float16* __restrict__ Qh,
                                            const _Float16* __restrict__ Kh,
                                            const _Float16* __restrict__ Vt,
                                            const int* __restrict__ amask,
                                            _Float16* __restrict__ Oh) {
    __shared__ _Float16 pls[8][16 * 32];   // per-wave 16x32 P staging tile (8 KB)

    const int lane = threadIdx.x & 31;
    const int wave = threadIdx.x >> 5;
    const int b    = blockIdx.y >> 4;
    const int head = blockIdx.y & 15;
    const int q0   = blockIdx.x * 128 + wave * 16;

    const _Float16* Qp = Qh + ((size_t)b * HH + head) * NN * HD;
    const _Float16* Kp = Kh + ((size_t)b * HH + head) * NN * HD;
    const _Float16* Vp = Vt + ((size_t)b * HH + head) * HD * NN; // [HD][NN]
    const int* mp = amask + b * NN;

    // Q as two A fragments (K-dim 64 = 2 x 32)
    v16h aq0 = load_afrag(Qp + (size_t)q0 * HD +  0, HD, lane);
    v16h aq1 = load_afrag(Qp + (size_t)q0 * HD + 32, HD, lane);

    v8f zero = {0.f,0.f,0.f,0.f,0.f,0.f,0.f,0.f};
    v8f o[4] = {zero, zero, zero, zero};
    float mrow[8], lrow[8];
#pragma unroll
    for (int r = 0; r < 8; ++r) { mrow[r] = -1e30f; lrow[r] = 0.f; }

    const float negInf = -__builtin_inff();
    _Float16* pw = &pls[wave][0];

    v16h kf[2][4];
    auto loadK = [&](int j, v16h* d) {
        d[0] = load_bfrag_rm(Kp + (size_t)j * HD,             HD, lane);
        d[1] = load_bfrag_rm(Kp + (size_t)j * HD + 32,        HD, lane);
        d[2] = load_bfrag_rm(Kp + (size_t)(j + 16) * HD,      HD, lane);
        d[3] = load_bfrag_rm(Kp + (size_t)(j + 16) * HD + 32, HD, lane);
    };

    auto step = [&](int j, const v16h* k4) {
        // --- S = Q K^T (two 16x16 tiles covering 32 keys) ---
        v8f s0 = wmma_f16(aq1, k4[1], wmma_f16(aq0, k4[0], zero));
        v8f s1 = wmma_f16(aq1, k4[3], wmma_f16(aq0, k4[2], zero));

        // V fragments for this block, issued before the softmax VALU section
        v16h bv[4];
#pragma unroll
        for (int t = 0; t < 4; ++t)
            bv[t] = load_bfrag_rm(Vp + (size_t)(t * 16) * NN + j, NN, lane);
        float mk0 = mp[j      + (lane & 15)] ? 0.f : negInf;
        float mk1 = mp[j + 16 + (lane & 15)] ? 0.f : negInf;

        // --- online softmax row stats (rows live in 16-lane halves) ---
#pragma unroll
        for (int r = 0; r < 8; ++r) {
            float e0 = s0[r] * SCALE + mk0;
            float e1 = s1[r] * SCALE + mk1;
            float mx = fmaxf(e0, e1);
            mx = fmaxf(mx, __shfl_xor(mx, 1, 32));
            mx = fmaxf(mx, __shfl_xor(mx, 2, 32));
            mx = fmaxf(mx, __shfl_xor(mx, 4, 32));
            mx = fmaxf(mx, __shfl_xor(mx, 8, 32));
            float mnew = fmaxf(fmaxf(mrow[r], mx), -1e30f); // clamp: all-masked safe
            float alpha = __expf(mrow[r] - mnew);
            mrow[r] = mnew;
            float p0 = __expf(e0 - mnew);   // -inf -> 0
            float p1 = __expf(e1 - mnew);
            float rs = p0 + p1;
            rs += __shfl_xor(rs, 1, 32);
            rs += __shfl_xor(rs, 2, 32);
            rs += __shfl_xor(rs, 4, 32);
            rs += __shfl_xor(rs, 8, 32);
            lrow[r] = lrow[r] * alpha + rs;
#pragma unroll
            for (int t = 0; t < 4; ++t) o[t][r] *= alpha;
            s0[r] = p0; s1[r] = p1;         // reuse as P
        }

        // --- C-layout -> A-layout via private LDS tile (same wave only) ---
#pragma unroll
        for (int r = 0; r < 8; ++r) {
            int M = (lane < 16) ? r : 8 + r;
            pw[M * 32 +      (lane & 15)] = (_Float16)s0[r];
            pw[M * 32 + 16 + (lane & 15)] = (_Float16)s1[r];
        }
        v16h ap = load_afrag(pw, 32, lane);   // P as A fragment (16x32)

        // --- O += P V  (V transposed: row d, contiguous over keys) ---
#pragma unroll
        for (int t = 0; t < 4; ++t) o[t] = wmma_f16(ap, bv[t], o[t]);
    };

    loadK(0, kf[0]);
    for (int j = 0; j < NN; j += 64) {
        loadK(j + 32, kf[1]);
        step(j, kf[0]);
        loadK((j + 64) & (NN - 1), kf[0]);   // wrap keeps last iteration defined
        step(j + 32, kf[1]);
    }

    // --- epilogue: normalize and store heads concatenated: [b][n][D] f16 ---
#pragma unroll
    for (int r = 0; r < 8; ++r) {
        float inv = (lrow[r] > 0.f) ? 1.f / lrow[r] : 0.f;  // all-masked -> 0
        int M = (lane < 16) ? r : 8 + r;
        int n = q0 + M;
#pragma unroll
        for (int t = 0; t < 4; ++t) {
            Oh[((size_t)b * NN + n) * DD + head * 64 + t * 16 + (lane & 15)] =
                (_Float16)(o[t][r] * inv);
        }
    }
}

// ---------------- kernel 5: output projection (f32 out) ----------------
// grid: x = BN/256, y = DD/64. Same ping-pong 32x64 tile as qkv_rope.
__global__ __launch_bounds__(256) void out_proj(const _Float16* __restrict__ Oh,
                                                const _Float16* __restrict__ wp,
                                                float* __restrict__ out) {
    const int lane = threadIdx.x & 31;
    const int wave = threadIdx.x >> 5;
    const int row0 = blockIdx.x * 256 + wave * 32;
    const int nt0  = blockIdx.y * 4;

    const _Float16* ab = Oh + (size_t)row0 * DD;
    const _Float16* wb = wp + ((size_t)3 * 2048 + nt0) * 512 + lane * 16;

    v8f zero = {0.f,0.f,0.f,0.f,0.f,0.f,0.f,0.f};
    v8f c[2][4] = {{zero,zero,zero,zero},{zero,zero,zero,zero}};

    v16h a0[2], a1[2], b[2][4];
    a0[0] = load_afrag(ab, DD, lane);
    a1[0] = load_afrag(ab + 16 * DD, DD, lane);
#pragma unroll
    for (int t = 0; t < 4; ++t) b[0][t] = *(const v16h*)(wb + (size_t)t * 512);

    for (int kt = 0; kt < DD / 32; kt += 2) {
        a0[1] = load_afrag(ab + (kt + 1) * 32, DD, lane);
        a1[1] = load_afrag(ab + 16 * DD + (kt + 1) * 32, DD, lane);
#pragma unroll
        for (int t = 0; t < 4; ++t)
            b[1][t] = *(const v16h*)(wb + ((size_t)(kt + 1) * 64 + t) * 512);
#pragma unroll
        for (int t = 0; t < 4; ++t) {
            c[0][t] = wmma_f16(a0[0], b[0][t], c[0][t]);
            c[1][t] = wmma_f16(a1[0], b[0][t], c[1][t]);
        }
        const int k2 = (kt + 2) & 31;
        a0[0] = load_afrag(ab + k2 * 32, DD, lane);
        a1[0] = load_afrag(ab + 16 * DD + k2 * 32, DD, lane);
#pragma unroll
        for (int t = 0; t < 4; ++t)
            b[0][t] = *(const v16h*)(wb + ((size_t)k2 * 64 + t) * 512);
#pragma unroll
        for (int t = 0; t < 4; ++t) {
            c[0][t] = wmma_f16(a0[1], b[1][t], c[0][t]);
            c[1][t] = wmma_f16(a1[1], b[1][t], c[1][t]);
        }
    }

#pragma unroll
    for (int mt = 0; mt < 2; ++mt) {
#pragma unroll
        for (int r = 0; r < 8; ++r) {
            int M = mt * 16 + ((lane < 16) ? r : 8 + r);
#pragma unroll
            for (int t = 0; t < 4; ++t)
                out[(size_t)(row0 + M) * DD + nt0 * 16 + t * 16 + (lane & 15)] =
                    c[mt][t][r];
        }
    }
}

// ---------------- launcher ----------------
extern "C" void kernel_launch(void* const* d_in, const int* in_sizes, int n_in,
                              void* d_out, int out_size, void* d_ws, size_t ws_size,
                              hipStream_t stream) {
    const float* x    = (const float*)d_in[0];
    const int*   mask = (const int*)  d_in[1];
    const float* Wq   = (const float*)d_in[2];
    const float* Wk   = (const float*)d_in[3];
    const float* Wv   = (const float*)d_in[4];
    const float* Wo   = (const float*)d_in[5];
    float* out = (float*)d_out;

    // workspace carve (48 MiB total, all f16)
    const size_t MB = 1ull << 20;
    char* w = (char*)d_ws;
    _Float16* xh = (_Float16*)(w +  0 * MB);  // [BN][DD]
    _Float16* wp = (_Float16*)(w +  8 * MB);  // packed weights, 4 mats
    _Float16* Qh = (_Float16*)(w + 16 * MB);  // [B][H][N][hd]
    _Float16* Kh = (_Float16*)(w + 24 * MB);  // [B][H][N][hd]
    _Float16* Vt = (_Float16*)(w + 32 * MB);  // [B][H][hd][N]
    _Float16* Oh = (_Float16*)(w + 40 * MB);  // [B][N][D]

    cvt_x   <<<dim3((BN * DD) / 1024),     256, 0, stream>>>(x, xh);
    pack_w  <<<dim3((4 * DD * DD) / 256),  256, 0, stream>>>(Wq, Wk, Wv, Wo, wp);
    qkv_rope<<<dim3(BN / 256, 3 * HH),     256, 0, stream>>>(xh, wp, Qh, Kh, Vt);
    attn    <<<dim3(NN / 128, Bz * HH),    256, 0, stream>>>(Qh, Kh, Vt, mask, Oh);
    out_proj<<<dim3(BN / 256, DD / 64),    256, 0, stream>>>(Oh, wp, out);
}
